// MultiHeadAttention_12627203850645
// MI455X (gfx1250) — matrix operand order repaired and checked by reading, not compiled, not run on hardware
//
#include <hip/hip_runtime.h>
#include <hip/hip_bf16.h>
#include <math.h>

#define BB  8
#define SS  1024
#define HH  1024
#define NHH 16
#define DKK 64
#define MM  (BB*SS)   // 8192 rows

typedef __attribute__((ext_vector_type(16))) _Float16 v16h;
typedef __attribute__((ext_vector_type(8)))  _Float16 v8h;
typedef __attribute__((ext_vector_type(8)))  float    v8f;

static __device__ __forceinline__ v8f wmma16(v16h a, v16h b, v8f c) {
  // D = A(16x32 f16) * B(32x16 f16) + C(16x16 f32)
  return __builtin_amdgcn_wmma_f32_16x16x32_f16(false, a, false, b, (short)0, c, false, false);
}
static __device__ __forceinline__ v16h cat8(v8h lo, v8h hi) {
  return __builtin_shufflevector(lo, hi, 0,1,2,3,4,5,6,7,8,9,10,11,12,13,14,15);
}
// A 16x32 fragment: lane row = lane&15; elems 0..15 = K base+16*half .. base+16*half+15
static __device__ __forceinline__ v16h load_a_frag(const _Float16* row, int halfsel, int k0) {
  const _Float16* p = row + k0 + halfsel * 16;
  return cat8(*(const v8h*)p, *(const v8h*)(p + 8));
}
// B 32x16 fragments: lane col = lane&15 (row of W); lanes 0-15: K 0-15, lanes 16-31: K 16-31
static __device__ __forceinline__ void load_b_frags(v16h (&b)[4], const _Float16* Wcol, int k0) {
  b[0] = *(const v16h*)(Wcol + k0);
  b[1] = *(const v16h*)(Wcol + k0 + 16 * HH);
  b[2] = *(const v16h*)(Wcol + k0 + 32 * HH);
  b[3] = *(const v16h*)(Wcol + k0 + 48 * HH);
}

// ---------------- f32 -> f16 convert ----------------
__global__ void cvt_f32_f16(const float* __restrict__ src, _Float16* __restrict__ dst, int n) {
  int i = blockIdx.x * blockDim.x + threadIdx.x;
  int stride = gridDim.x * blockDim.x;
  for (; i < n; i += stride) dst[i] = (_Float16)src[i];
}

// ---------------- GEMM: Y[M,1024] = X[M,1024] @ W[1024,1024]^T ----------------
// Block = 8 waves sharing one 16-row A strip, staged to LDS via CDNA5 async copy.
// Each wave owns a 16x64 output strip; 2-deep software pipeline on K.
// mode 0: f16 row-major (Q,K)  mode 1: f16 transposed [B,NH,DK,S] (V^T)  mode 2: f32 (O-proj)
__global__ void __launch_bounds__(256) gemm_xwt(
    const _Float16* __restrict__ X, const _Float16* __restrict__ W,
    _Float16* __restrict__ out16, float* __restrict__ out32, int mode) {
  __shared__ _Float16 As[16 * HH];   // 32 KB A strip shared by the block
  const int tid  = threadIdx.x;
  const int gw0  = blockIdx.x * 8;         // first global wave id of this block
  const int m0   = (gw0 >> 4) << 4;        // same for all 8 waves in the block
  const int wv   = tid >> 5;
  const int lane = tid & 31;
  const int halfsel = lane >> 4;
  const int lr   = lane & 15;
  const int n0   = ((gw0 + wv) & 15) << 6;

  // --- async stage A strip (16 x 1024 f16 = 32KB) into LDS: 2048 b128 chunks ---
  {
    const _Float16* src = X + (size_t)m0 * HH;
    const unsigned ldsbase = (unsigned)(size_t)(&As[0]);
#pragma unroll
    for (int t = 0; t < 8; t++) {
      const int idx = tid + t * 256;                 // 16-byte chunk id
      const unsigned lds_addr = ldsbase + idx * 16;  // LDS byte offset (VDST)
      const _Float16* g = src + idx * 8;             // global address (VADDR)
      asm volatile("global_load_async_to_lds_b128 %0, %1, off"
                   :: "v"(lds_addr), "v"(g) : "memory");
    }
    asm volatile("s_wait_asynccnt 0x0" ::: "memory");
  }
  __syncthreads();

  const _Float16* Arow = As + lr * HH;
  const _Float16* Wcol = W + (size_t)(n0 + lr) * HH + halfsel * 16;

  v8f acc[4] = {v8f{}, v8f{}, v8f{}, v8f{}};
  v16h aP, aN, bP[4], bN[4];
  aP = load_a_frag(Arow, halfsel, 0);
  load_b_frags(bP, Wcol, 0);
  for (int k0 = 0; k0 < HH; k0 += 64) {
    // prefetch K-step k0+32 while computing k0
    aN = load_a_frag(Arow, halfsel, k0 + 32);
    load_b_frags(bN, Wcol, k0 + 32);
    acc[0] = wmma16(aP, bP[0], acc[0]);
    acc[1] = wmma16(aP, bP[1], acc[1]);
    acc[2] = wmma16(aP, bP[2], acc[2]);
    acc[3] = wmma16(aP, bP[3], acc[3]);
    // prefetch K-step k0+64 (clamped on last iter; result unused but in-bounds)
    const int kn = (k0 + 64 < HH) ? (k0 + 64) : 0;
    aP = load_a_frag(Arow, halfsel, kn);
    load_b_frags(bP, Wcol, kn);
    acc[0] = wmma16(aN, bN[0], acc[0]);
    acc[1] = wmma16(aN, bN[1], acc[1]);
    acc[2] = wmma16(aN, bN[2], acc[2]);
    acc[3] = wmma16(aN, bN[3], acc[3]);
  }

#pragma unroll
  for (int j = 0; j < 4; j++) {
#pragma unroll
    for (int i = 0; i < 8; i++) {
      const int r = i + halfsel * 8;     // D: row = i + 8*half, col = lane&15
      const int m = m0 + r;
      const int c = n0 + j * 16 + lr;
      const float v = acc[j][i];
      if (mode == 0) {
        out16[(size_t)m * HH + c] = (_Float16)v;
      } else if (mode == 1) {
        const int b_ = m >> 10, s_ = m & 1023;
        const int h_ = c >> 6,  d_ = c & 63;
        out16[(((size_t)b_ * NHH + h_) * DKK + d_) * SS + s_] = (_Float16)v;
      } else {
        out32[(size_t)m * HH + c] = v;
      }
    }
  }
}

// ---------------- Fused attention: scores -> mask -> softmax -> attn_dist + attn@V ----------------
// grid (S/16, NH, B), 256 threads (8 waves). Stage 1: wave w owns key tiles {w, w+8, ..., w+56}.
__global__ void __launch_bounds__(256) attn_kernel(
    const _Float16* __restrict__ Qh, const _Float16* __restrict__ Kh,
    const _Float16* __restrict__ Vth, const int* __restrict__ mask,
    float* __restrict__ attn_out, _Float16* __restrict__ AOh) {
  const int b  = blockIdx.z, h = blockIdx.y;
  const int q0 = blockIdx.x << 4;
  const int tid = threadIdx.x;
  const int wave = tid >> 5, lane = tid & 31;
  const int halfsel = lane >> 4, lr = lane & 15;

  __shared__ _Float16 a16[16 * SS];   // 32 KB: normalized probs (f16) for attn@V
  __shared__ float red[8][16];
  __shared__ float rowstat[16];

  // A fragments of Q tile (rows q0..q0+15, d = 0..63), reused for all 64 key tiles
  const _Float16* Qrow = Qh + (size_t)(b * SS + q0 + lr) * HH + h * DKK;
  v16h aq0 = load_a_frag(Qrow, halfsel, 0);
  v16h aq1 = load_a_frag(Qrow, halfsel, 32);

  float vals[8][8];
  float pr[8];
#pragma unroll
  for (int i = 0; i < 8; i++) pr[i] = -3.0e38f;

#pragma unroll
  for (int j = 0; j < 8; j++) {
    const int k0 = (wave + (j << 3)) << 4;
    const _Float16* Kp = Kh + (size_t)(b * SS + k0 + lr) * HH + h * DKK + halfsel * 16;
    v16h bk0 = *(const v16h*)(Kp);        // B column n = K row k0+n (d = reduction dim)
    v16h bk1 = *(const v16h*)(Kp + 32);
    v8f c = {};
    c = wmma16(aq0, bk0, c);
    c = wmma16(aq1, bk1, c);
#pragma unroll
    for (int i = 0; i < 8; i++) {
      const int sr = q0 + i + halfsel * 8;
      const int sc = k0 + lr;
      float v = c[i] * 0.125f;                                  // 1/sqrt(64)
      if (mask[(size_t)(b * SS + sr) * SS + sc] == 0) v = -1.0e9f;
      vals[j][i] = v;
      pr[i] = fmaxf(pr[i], v);
    }
  }
  // row max: reduce across the 16 lanes of each half (same rows), then across waves via LDS
#pragma unroll
  for (int off = 1; off < 16; off <<= 1)
#pragma unroll
    for (int i = 0; i < 8; i++) pr[i] = fmaxf(pr[i], __shfl_xor(pr[i], off, 32));
  if (lr == 0)
#pragma unroll
    for (int i = 0; i < 8; i++) red[wave][i + halfsel * 8] = pr[i];
  __syncthreads();
  if (tid < 16) {
    float m = -3.0e38f;
    for (int w = 0; w < 8; w++) m = fmaxf(m, red[w][tid]);
    rowstat[tid] = m;
  }
  __syncthreads();

  float ps[8];
#pragma unroll
  for (int i = 0; i < 8; i++) ps[i] = 0.f;
#pragma unroll
  for (int j = 0; j < 8; j++)
#pragma unroll
    for (int i = 0; i < 8; i++) {
      float e = __expf(vals[j][i] - rowstat[i + halfsel * 8]);
      vals[j][i] = e;
      ps[i] += e;
    }
#pragma unroll
  for (int off = 1; off < 16; off <<= 1)
#pragma unroll
    for (int i = 0; i < 8; i++) ps[i] += __shfl_xor(ps[i], off, 32);
  if (lr == 0)
#pragma unroll
    for (int i = 0; i < 8; i++) red[wave][i + halfsel * 8] = ps[i];
  __syncthreads();
  if (tid < 16) {
    float s = 0.f;
    for (int w = 0; w < 8; w++) s += red[w][tid];
    rowstat[tid] = 1.0f / s;
  }
  __syncthreads();

  // normalize; write attn_dist once (f32, streaming/non-temporal) + f16 copy to LDS
  float* aout = attn_out + (size_t)(b * NHH + h) * SS * SS;
#pragma unroll
  for (int j = 0; j < 8; j++) {
    const int k0 = (wave + (j << 3)) << 4;
#pragma unroll
    for (int i = 0; i < 8; i++) {
      const int r = i + halfsel * 8;
      const float a = vals[j][i] * rowstat[r];
      __builtin_nontemporal_store(a, &aout[(size_t)(q0 + r) * SS + k0 + lr]);
      a16[r * SS + k0 + lr] = (_Float16)a;
    }
  }
  __syncthreads();

  // Stage 2: out[16, 64] = attn[16,1024] @ V[1024,64]; waves 0..3 each own a 16-col d-tile
  if (wave < 4) {
    const int d0 = wave << 4;
    v8f acc = {};
    const _Float16* Vp = Vth + (size_t)((b * NHH + h) * DKK + d0 + lr) * SS; // V^T: contiguous in k
    for (int kk = 0; kk < SS; kk += 32) {
      const _Float16* ap = a16 + lr * SS + kk + halfsel * 16;
      v16h af = cat8(*(const v8h*)ap, *(const v8h*)(ap + 8));
      v16h bf = *(const v16h*)(Vp + kk + halfsel * 16);
      acc = wmma16(af, bf, acc);
    }
#pragma unroll
    for (int i = 0; i < 8; i++) {
      const int r = i + halfsel * 8;
      AOh[(size_t)(b * SS + q0 + r) * HH + h * DKK + d0 + lr] = (_Float16)acc[i];
    }
  }
}

// ---------------- residual + LayerNorm ----------------
__global__ void __launch_bounds__(256) ln_residual(
    const float* __restrict__ proj, const float* __restrict__ enc,
    const float* __restrict__ gamma, const float* __restrict__ beta,
    float* __restrict__ out) {
  const int row = blockIdx.x;
  const int tid = threadIdx.x;
  __shared__ float sbuf[256];
  const float* p = proj + (size_t)row * HH;
  const float* e = enc  + (size_t)row * HH;
  float x[4];
  float s = 0.f;
#pragma unroll
  for (int k = 0; k < 4; k++) { x[k] = p[tid + k * 256] + e[tid + k * 256]; s += x[k]; }
  sbuf[tid] = s; __syncthreads();
  for (int off = 128; off > 0; off >>= 1) { if (tid < off) sbuf[tid] += sbuf[tid + off]; __syncthreads(); }
  const float mu = sbuf[0] * (1.0f / HH);
  __syncthreads();
  float v = 0.f;
#pragma unroll
  for (int k = 0; k < 4; k++) { const float d = x[k] - mu; v += d * d; }
  sbuf[tid] = v; __syncthreads();
  for (int off = 128; off > 0; off >>= 1) { if (tid < off) sbuf[tid] += sbuf[tid + off]; __syncthreads(); }
  const float rstd = rsqrtf(sbuf[0] * (1.0f / HH) + 1e-6f);
#pragma unroll
  for (int k = 0; k < 4; k++) {
    const int c = tid + k * 256;
    const float y = (x[k] - mu) * rstd * gamma[c] + beta[c];
    __builtin_nontemporal_store(y, &out[(size_t)row * HH + c]);
  }
}

extern "C" void kernel_launch(void* const* d_in, const int* in_sizes, int n_in,
                              void* d_out, int out_size, void* d_ws, size_t ws_size,
                              hipStream_t stream) {
  (void)in_sizes; (void)n_in; (void)out_size; (void)ws_size;
  const float* enc   = (const float*)d_in[0];
  const int*   mask  = (const int*)  d_in[1];
  const float* W_Q   = (const float*)d_in[2];
  const float* W_K   = (const float*)d_in[3];
  const float* W_V   = (const float*)d_in[4];
  const float* W_O   = (const float*)d_in[5];
  const float* gamma = (const float*)d_in[6];
  const float* beta  = (const float*)d_in[7];
  float* out = (float*)d_out;

  char* ws = (char*)d_ws;
  size_t off = 0;
  _Float16* ench = (_Float16*)(ws + off); off += (size_t)MM * HH * 2;   // 16 MB
  _Float16* wqh  = (_Float16*)(ws + off); off += (size_t)HH * HH * 2;
  _Float16* wkh  = (_Float16*)(ws + off); off += (size_t)HH * HH * 2;
  _Float16* wvh  = (_Float16*)(ws + off); off += (size_t)HH * HH * 2;
  _Float16* woh  = (_Float16*)(ws + off); off += (size_t)HH * HH * 2;
  const size_t offQ = off;
  _Float16* Qh  = (_Float16*)(ws + off); off += (size_t)MM * HH * 2;    // 16 MB
  _Float16* Kh  = (_Float16*)(ws + off); off += (size_t)MM * HH * 2;    // 16 MB
  _Float16* Vth = (_Float16*)(ws + off); off += (size_t)MM * HH * 2;    // 16 MB
  _Float16* AOh = (_Float16*)(ws + off); off += (size_t)MM * HH * 2;    // 16 MB
  float* proj = (float*)(ws + offQ);  // 32 MB, aliases Q+K (dead after attention)

  cvt_f32_f16<<<1024, 256, 0, stream>>>(enc, ench, MM * HH);
  cvt_f32_f16<<<256, 256, 0, stream>>>(W_Q, wqh, HH * HH);
  cvt_f32_f16<<<256, 256, 0, stream>>>(W_K, wkh, HH * HH);
  cvt_f32_f16<<<256, 256, 0, stream>>>(W_V, wvh, HH * HH);
  cvt_f32_f16<<<256, 256, 0, stream>>>(W_O, woh, HH * HH);

  gemm_xwt<<<1024, 256, 0, stream>>>(ench, wqh, Qh,  nullptr, 0);
  gemm_xwt<<<1024, 256, 0, stream>>>(ench, wkh, Kh,  nullptr, 0);
  gemm_xwt<<<1024, 256, 0, stream>>>(ench, wvh, Vth, nullptr, 1);

  attn_kernel<<<dim3(SS / 16, NHH, BB), 256, 0, stream>>>(
      Qh, Kh, Vth, mask, out + (size_t)MM * HH, AOh);

  gemm_xwt<<<1024, 256, 0, stream>>>(AOh, woh, nullptr, proj, 2);
  ln_residual<<<MM, 256, 0, stream>>>(proj, enc, gamma, beta, out);
}
